// MultiHeadCrossAttention_1382979469929
// MI455X (gfx1250) — matrix operand order repaired and checked by reading, not compiled, not run on hardware
//
#include <hip/hip_runtime.h>
#include <hip/hip_bf16.h>
#include <stdint.h>

// Shapes (fixed by the reference)
#define B_  4
#define S_  2048   // N == M
#define C_  1024
#define H_  16
#define D_  64

typedef __attribute__((ext_vector_type(16))) _Float16 v16h;
typedef __attribute__((ext_vector_type(8)))  float    v8f;
typedef __attribute__((ext_vector_type(4)))  uint32_t u32x4;
typedef __attribute__((ext_vector_type(8)))  int      i32x8;
typedef __attribute__((ext_vector_type(4)))  int      i32x4;

union Frag16 {
  v16h     v;
  _Float16 h[16];
  uint32_t u[8];
};
union Pack8 {
  _Float16 h[8];
  u32x4    q;
};

__device__ __forceinline__ v8f wmma_f16(const Frag16& a, const Frag16& b, v8f c) {
  // D = A(16x32 f16) * B(32x16 f16) + C(16x16 f32)
  return __builtin_amdgcn_wmma_f32_16x16x32_f16(false, a.v, false, b.v,
                                                (short)0, c, false, false);
}

// ---------------------------------------------------------------------------
// Tensor Data Mover: stage a (tile_d1 rows x tile_d0 f16) tile into LDS with
// TDM row padding (16B pad per row, pad_code selects row payload 64B/128B)
// so the DS B128 reads below are bank-conflict free.
// ---------------------------------------------------------------------------
#if __has_builtin(__builtin_amdgcn_tensor_load_to_lds) && \
    __has_builtin(__builtin_amdgcn_s_wait_tensorcnt)
#define USE_TDM 1
#endif

#if __has_include(<hip/amd_detail/amd_gfx1250_TDM.h>)
#define TDM_6ARG 1   // amdgpu-toolchain (clang-23/therock): 6-arg builtin
#endif

#ifdef USE_TDM
// pad_code 3: pad after 16 dwords (64B rows) ; pad_code 4: after 32 dwords (128B)
__device__ __forceinline__ void tdm_load_tile(const _Float16* gsrc,
                                              uint32_t lds_byte_addr,
                                              int tile_d0, int tile_d1,
                                              int stride_d0, int pad_code) {
  const uint64_t ga = (uint64_t)(uintptr_t)gsrc;
  // D# group 0: count=1 descriptor, lds addr, global addr, type=2 ("image")
  u32x4 g0;
  g0[0] = 1u;
  g0[1] = lds_byte_addr;
  g0[2] = (uint32_t)ga;
  g0[3] = (uint32_t)((ga >> 32) & 0x1FFFFFFu) | (2u << 30);
  // D# group 1: data_size=1(2B) | pad_enable | pad_interval | pad_amount=4dw
  i32x8 g1 = {};
  g1[0] = (1 << 16) | (1 << 20) | (pad_code << 22) | (3 << 25);
  const uint32_t T0 = 1u << 30, T1 = 1u << 30;          // generous OOB dims
  g1[1] = (int)((T0 & 0xFFFFu) << 16);                  // tensor_dim0 lo
  g1[2] = (int)(((T0 >> 16) & 0xFFFFu) | ((T1 & 0xFFFFu) << 16));
  g1[3] = (int)(((T1 >> 16) & 0xFFFFu) | ((uint32_t)tile_d0 << 16));
  g1[4] = tile_d1;
  g1[5] = stride_d0;
  const i32x4 z4 = {};
#ifdef TDM_6ARG
  const i32x8 z8 = {};
  __builtin_amdgcn_tensor_load_to_lds(g0, g1, z4, z4, z8, 0);
#else
  __builtin_amdgcn_tensor_load_to_lds(g0, g1, z4, z4, 0);
#endif
}
#define TDM_WAIT() __builtin_amdgcn_s_wait_tensorcnt(0)
#endif

// ---------------------------------------------------------------------------
// GEMM core: acc[4] (16x64 tile) += X_tile(16xC) * W(colBase..colBase+63, :)^T
// All 8 waves of the block share colBase; the W tile is TDM-staged once,
// double-buffered ahead of compute.
// ---------------------------------------------------------------------------
template <bool A_IS_F32>
__device__ __forceinline__ void gemm_core(const void* __restrict__ Xrow,
                                          const _Float16* __restrict__ W,
                                          int colBase, int l16, int hi,
                                          v8f acc[4]) {
#ifdef USE_TDM
  __shared__ _Float16 wbuf[2][64][40];   // 64B payload + 16B TDM pad per row
  if ((threadIdx.x >> 5) == 0)
    tdm_load_tile(W + (size_t)colBase * C_,
                  (uint32_t)(uintptr_t)&wbuf[0][0][0], 32, 64, C_, 3);
  int cur = 0;
#endif
  for (int kk = 0; kk < C_; kk += 32) {
#ifdef USE_TDM
    TDM_WAIT();             // scalar; non-issuing waves have TENSORcnt==0
    __syncthreads();
    if (kk + 32 < C_ && (threadIdx.x >> 5) == 0)
      tdm_load_tile(W + (size_t)colBase * C_ + (kk + 32),
                    (uint32_t)(uintptr_t)&wbuf[cur ^ 1][0][0], 32, 64, C_, 3);
#endif
    Frag16 a;
    if constexpr (A_IS_F32) {
      const float* xr = (const float*)Xrow;
      if (kk + 128 < C_) __builtin_prefetch(xr + kk + 128, 0, 1);
#pragma unroll
      for (int v = 0; v < 8; ++v) {
        const int k0 = kk + 2 * v + (v >= 4 ? 8 : 0) + 8 * hi;
        const float2 f = *(const float2*)(xr + k0);
        a.h[2 * v]     = (_Float16)f.x;
        a.h[2 * v + 1] = (_Float16)f.y;
      }
    } else {
      const _Float16* xr = (const _Float16*)Xrow;
      if (kk + 256 < C_) __builtin_prefetch(xr + kk + 256, 0, 1);
#pragma unroll
      for (int v = 0; v < 8; ++v) {
        const int k0 = kk + 2 * v + (v >= 4 ? 8 : 0) + 8 * hi;
        a.u[v] = *(const uint32_t*)(xr + k0);
      }
    }
    // Preload all 4 B-frags, then burst the WMMAs (loads overlap waits).
    Frag16 bf[4];
#pragma unroll
    for (int t = 0; t < 4; ++t) {
#ifdef USE_TDM
      const _Float16* wr = &wbuf[cur][t * 16 + l16][0];
#pragma unroll
      for (int j = 0; j < 8; ++j)
        bf[t].u[j] = *(const uint32_t*)(wr + 2 * j + (j >= 4 ? 8 : 0) + 8 * hi);
#else
      const _Float16* wrow = W + (size_t)(colBase + t * 16 + l16) * C_ + kk;
#pragma unroll
      for (int j = 0; j < 8; ++j)
        bf[t].u[j] = *(const uint32_t*)(wrow + 2 * j + (j >= 4 ? 8 : 0) + 8 * hi);
#endif
    }
#pragma unroll
    for (int t = 0; t < 4; ++t) acc[t] = wmma_f16(a, bf[t], acc[t]);
#ifdef USE_TDM
    cur ^= 1;
#endif
  }
}

// ---------------------------------------------------------------------------
// f32 -> f16 elementwise convert (for weights)
// ---------------------------------------------------------------------------
__global__ void cvt_f32_f16_kernel(const float* __restrict__ in,
                                   _Float16* __restrict__ out, int n) {
  int i = blockIdx.x * blockDim.x + threadIdx.x;
  if (i < n) out[i] = (_Float16)in[i];
}

// ---------------------------------------------------------------------------
// Q/K projection: Out[b,h,s,d] = sum_k X[(b,s),k]*W[c,k] + bias[c], c=h*64+d
// ---------------------------------------------------------------------------
__global__ void proj_qk_kernel(const float* __restrict__ X,     // [B*S, C] f32
                               const _Float16* __restrict__ W,  // [C, C] f16
                               const float* __restrict__ bias,  // [C]
                               _Float16* __restrict__ Out)      // [B, H, S, D]
{
  const int lane = threadIdx.x & 31;
  const int l16  = lane & 15;
  const int hi   = lane >> 4;
  const int colGroup = blockIdx.x % (C_ / 64);
  const int rowBlock = blockIdx.x / (C_ / 64);
  const int rowBase  = (rowBlock * 8 + (int)(threadIdx.x >> 5)) * 16;
  const int colBase  = colGroup * 64;

  v8f acc[4] = {};
  gemm_core<true>(X + (size_t)(rowBase + l16) * C_, W, colBase, l16, hi, acc);

#pragma unroll
  for (int t = 0; t < 4; ++t) {
    const int col = colBase + t * 16 + l16;
    const float bv = bias[col];
    const int h = col >> 6, d = col & 63;
#pragma unroll
    for (int r = 0; r < 8; ++r) {
      const int row = rowBase + r + 8 * hi;
      const int b = row >> 11;
      const int n = row & (S_ - 1);
      Out[(((size_t)(b * H_ + h) * S_ + n) * D_) + d] = (_Float16)(acc[t][r] + bv);
    }
  }
}

// ---------------------------------------------------------------------------
// V projection with TRANSPOSED output: Vt[b,h,d,s] (packed 16B stores; gives
// contiguous m-runs for the attention P*V B-operand and for TDM staging)
// ---------------------------------------------------------------------------
__global__ void proj_v_kernel(const float* __restrict__ X,     // [B*S, C] f32
                              const _Float16* __restrict__ W,  // [C, C] f16
                              const float* __restrict__ bias,  // [C]
                              _Float16* __restrict__ Vt)       // [B, H, D, S]
{
  const int lane = threadIdx.x & 31;
  const int l16  = lane & 15;
  const int hi   = lane >> 4;
  const int colGroup = blockIdx.x % (C_ / 64);
  const int rowBlock = blockIdx.x / (C_ / 64);
  const int rowBase  = (rowBlock * 8 + (int)(threadIdx.x >> 5)) * 16;
  const int colBase  = colGroup * 64;

  v8f acc[4] = {};
  gemm_core<true>(X + (size_t)(rowBase + l16) * C_, W, colBase, l16, hi, acc);

  const int row0 = rowBase + 8 * hi;     // 8 consecutive s-indices per lane
  const int b = row0 >> 11;
  const int n0 = row0 & (S_ - 1);
#pragma unroll
  for (int t = 0; t < 4; ++t) {
    const int col = colBase + t * 16 + l16;
    const float bv = bias[col];
    const int h = col >> 6, d = col & 63;
    Pack8 s;
#pragma unroll
    for (int r = 0; r < 8; ++r) s.h[r] = (_Float16)(acc[t][r] + bv);
    *(u32x4*)&Vt[((size_t)(b * H_ + h) * D_ + d) * S_ + n0] = s.q;
  }
}

// ---------------------------------------------------------------------------
// Output projection: Out[(b,s),c] = sum_k O16[(b,s),k]*Wo[c,k] + bo[c] (f32)
// ---------------------------------------------------------------------------
__global__ void proj_o_kernel(const _Float16* __restrict__ X,   // [B*S, C] f16
                              const _Float16* __restrict__ W,   // [C, C] f16
                              const float* __restrict__ bias,   // [C]
                              float* __restrict__ Out)          // [B*S, C] f32
{
  const int lane = threadIdx.x & 31;
  const int l16  = lane & 15;
  const int hi   = lane >> 4;
  const int colGroup = blockIdx.x % (C_ / 64);
  const int rowBlock = blockIdx.x / (C_ / 64);
  const int rowBase  = (rowBlock * 8 + (int)(threadIdx.x >> 5)) * 16;
  const int colBase  = colGroup * 64;

  v8f acc[4] = {};
  gemm_core<false>(X + (size_t)(rowBase + l16) * C_, W, colBase, l16, hi, acc);

#pragma unroll
  for (int t = 0; t < 4; ++t) {
    const int col = colBase + t * 16 + l16;
    const float bv = bias[col];
#pragma unroll
    for (int r = 0; r < 8; ++r) {
      const int row = rowBase + r + 8 * hi;
      Out[(size_t)row * C_ + col] = acc[t][r] + bv;
    }
  }
}

// ---------------------------------------------------------------------------
// Flash attention. Block = 8 waves sharing one (b,h); each wave owns one
// 16-row q-tile. K-tile (32x64) and V-tile (64 d-rows x 32 m, from Vt) are
// TDM-staged into LDS once per m-step (double-buffered) and broadcast-read
// by all 8 waves. Scores are computed TRANSPOSED (T = K*Q^T) so softmax over
// m is per-lane (+1 shfl) and exp(T) repacks into the WMMA A-operand layout
// entirely in-lane.
// ---------------------------------------------------------------------------
__global__ void attn_kernel(const _Float16* __restrict__ Q,   // [B*H, S, D]
                            const _Float16* __restrict__ K,   // [B*H, S, D]
                            const _Float16* __restrict__ Vt,  // [B*H, D, S]
                            _Float16* __restrict__ O)         // [B, S, C]
{
  const int lane = threadIdx.x & 31;
  const int l16  = lane & 15;
  const int hi   = lane >> 4;
  const int waveId = threadIdx.x >> 5;

  const int ntBlocks = (S_ / 16) / 8;          // 16
  const int bh = blockIdx.x / ntBlocks;        // 0..63 (shared by block)
  const int nt = (blockIdx.x % ntBlocks) * 8 + waveId;

  const _Float16* Qb  = Q  + (size_t)bh * S_ * D_;
  const _Float16* Kb  = K  + (size_t)bh * S_ * D_;
  const _Float16* Vtb = Vt + (size_t)bh * D_ * S_;

  // Q^T as B-operand: column n of B = Q row (nt*16 + n); two 32-wide d chunks.
  Frag16 qf[2];
  {
    const _Float16* qrow = Qb + (size_t)(nt * 16 + l16) * D_;
#pragma unroll
    for (int c = 0; c < 2; ++c)
#pragma unroll
      for (int j = 0; j < 8; ++j)
        qf[c].u[j] = *(const uint32_t*)(qrow + c * 32 + 2 * j + (j >= 4 ? 8 : 0) + 8 * hi);
  }

#ifdef USE_TDM
  __shared__ _Float16 kbuf[2][32][72];  // 128B payload + 16B TDM pad per row
  __shared__ _Float16 vbuf[2][64][40];  // 64B payload + 16B TDM pad per row
  if (waveId == 0) {
    tdm_load_tile(Kb, (uint32_t)(uintptr_t)&kbuf[0][0][0], 64, 32, D_, 4);
    tdm_load_tile(Vtb, (uint32_t)(uintptr_t)&vbuf[0][0][0], 32, 64, S_, 3);
  }
  int cur = 0;
#else
  const _Float16* vrow[4];
#pragma unroll
  for (int c = 0; c < 4; ++c) vrow[c] = Vtb + (size_t)(c * 16 + l16) * S_;
#endif

  const float scale = 0.125f;   // 1/sqrt(64)
  float mrun = -3.0e38f;
  float lsum = 0.0f;
  v8f o[4] = {};

  for (int mb = 0; mb < S_; mb += 32) {
#ifdef USE_TDM
    TDM_WAIT();
    __syncthreads();
    if (mb + 32 < S_ && waveId == 0) {
      tdm_load_tile(Kb + (size_t)(mb + 32) * D_,
                    (uint32_t)(uintptr_t)&kbuf[cur ^ 1][0][0], 64, 32, D_, 4);
      tdm_load_tile(Vtb + (mb + 32),
                    (uint32_t)(uintptr_t)&vbuf[cur ^ 1][0][0], 32, 64, S_, 3);
    }
#else
    if (mb + 32 < S_) {  // prefetch next tiles -> global_prefetch_b8
      __builtin_prefetch(Kb + (size_t)(mb + 32 + l16) * D_, 0, 1);
      __builtin_prefetch(Kb + (size_t)(mb + 48 + l16) * D_, 0, 1);
      __builtin_prefetch(vrow[0] + mb + 32, 0, 1);
      __builtin_prefetch(vrow[2] + mb + 32, 0, 1);
    }
#endif

    // ---- scores: T0 = rows m in [mb, mb+16), T1 = [mb+16, mb+32) ----
    Frag16 ka[4];   // [halfm*2 + dchunk]
#pragma unroll
    for (int halfm = 0; halfm < 2; ++halfm) {
#ifdef USE_TDM
      const _Float16* krow = &kbuf[cur][halfm * 16 + l16][0];
#else
      const _Float16* krow = Kb + (size_t)(mb + halfm * 16 + l16) * D_;
#endif
#pragma unroll
      for (int c = 0; c < 2; ++c)
#pragma unroll
        for (int v = 0; v < 8; ++v) {
          const int k0 = c * 32 + 2 * v + (v >= 4 ? 8 : 0) + 8 * hi;
          ka[halfm * 2 + c].u[v] = *(const uint32_t*)(krow + k0);
        }
    }
    v8f T0 = {}, T1 = {};
    T0 = wmma_f16(ka[0], qf[0], T0);
    T0 = wmma_f16(ka[1], qf[1], T0);
    T1 = wmma_f16(ka[2], qf[0], T1);
    T1 = wmma_f16(ka[3], qf[1], T1);

    // ---- streaming softmax over m (column n = l16 of T) ----
    float t0[8], t1[8];
    float smax = -3.0e38f;
#pragma unroll
    for (int r = 0; r < 8; ++r) {
      t0[r] = T0[r] * scale;
      t1[r] = T1[r] * scale;
      smax = fmaxf(smax, fmaxf(t0[r], t1[r]));
    }
    smax = fmaxf(smax, __shfl_xor(smax, 16, 32));
    const float mnew = fmaxf(mrun, smax);
    const float fac  = __expf(mrun - mnew);
    mrun = mnew;

    // P in WMMA A-layout: h[r] <- exp(T0[r]), h[8+r] <- exp(T1[r]) (in-lane!)
    Frag16 p;
    float psum = 0.0f;
#pragma unroll
    for (int r = 0; r < 8; ++r) {
      const float e0 = __expf(t0[r] - mnew);
      const float e1 = __expf(t1[r] - mnew);
      p.h[r]     = (_Float16)e0;
      p.h[8 + r] = (_Float16)e1;
      psum += e0 + e1;
    }
    lsum = lsum * fac + psum;

    // Rescale O accumulators: factor for row n = r + 8*hi lives in lane n.
    float frow[8];
#pragma unroll
    for (int r = 0; r < 8; ++r) frow[r] = __shfl(fac, r + 8 * hi, 32);
#pragma unroll
    for (int c = 0; c < 4; ++c)
#pragma unroll
      for (int r = 0; r < 8; ++r) o[c][r] *= frow[r];

    // ---- O += P(16x32) * V_tile(32x64), per 16-wide d chunk ----
    Frag16 vf[4];
#pragma unroll
    for (int c = 0; c < 4; ++c) {
#ifdef USE_TDM
      const _Float16* vr = &vbuf[cur][c * 16 + l16][0];
#pragma unroll
      for (int j = 0; j < 8; ++j)
        vf[c].u[j] = *(const uint32_t*)(vr + 2 * j + (j >= 4 ? 8 : 0) + 8 * hi);
#else
#pragma unroll
      for (int j = 0; j < 8; ++j)
        vf[c].u[j] = *(const uint32_t*)(vrow[c] + mb + 2 * j + (j >= 4 ? 8 : 0) + 8 * hi);
#endif
    }
#pragma unroll
    for (int c = 0; c < 4; ++c) o[c] = wmma_f16(p, vf[c], o[c]);

#ifdef USE_TDM
    cur ^= 1;
#endif
  }

  // Final normalization and store to [B, S, C] with c = h*64 + d.
  lsum += __shfl_xor(lsum, 16, 32);
  float rs[8];
#pragma unroll
  for (int r = 0; r < 8; ++r) rs[r] = 1.0f / __shfl(lsum, r + 8 * hi, 32);

  const int b = bh >> 4, h = bh & 15;
#pragma unroll
  for (int c = 0; c < 4; ++c) {
    const int d = c * 16 + l16;
#pragma unroll
    for (int r = 0; r < 8; ++r) {
      const int n = nt * 16 + r + 8 * hi;
      O[(size_t)(b * S_ + n) * C_ + h * D_ + d] = (_Float16)(o[c][r] * rs[r]);
    }
  }
}

// ---------------------------------------------------------------------------
// Host launcher
// ---------------------------------------------------------------------------
extern "C" void kernel_launch(void* const* d_in, const int* in_sizes, int n_in,
                              void* d_out, int out_size, void* d_ws, size_t ws_size,
                              hipStream_t stream) {
  const float* query = (const float*)d_in[0];
  const float* key   = (const float*)d_in[1];
  const float* value = (const float*)d_in[2];
  const float* Wq = (const float*)d_in[3];
  const float* bq = (const float*)d_in[4];
  const float* Wk = (const float*)d_in[5];
  const float* bk = (const float*)d_in[6];
  const float* Wv = (const float*)d_in[7];
  const float* bv = (const float*)d_in[8];
  const float* Wo = (const float*)d_in[9];
  const float* bo = (const float*)d_in[10];

  char* ws = (char*)d_ws;
  _Float16* Wq16 = (_Float16*)(ws + ((size_t)0  << 20));  // 2 MB
  _Float16* Wk16 = (_Float16*)(ws + ((size_t)2  << 20));  // 2 MB
  _Float16* Wv16 = (_Float16*)(ws + ((size_t)4  << 20));  // 2 MB
  _Float16* Wo16 = (_Float16*)(ws + ((size_t)6  << 20));  // 2 MB
  _Float16* Q16  = (_Float16*)(ws + ((size_t)8  << 20));  // 16 MB [B,H,S,D]
  _Float16* K16  = (_Float16*)(ws + ((size_t)24 << 20));  // 16 MB [B,H,S,D]
  _Float16* V16  = (_Float16*)(ws + ((size_t)40 << 20));  // 16 MB [B,H,D,S]
  _Float16* O16  = (_Float16*)(ws + ((size_t)56 << 20));  // 16 MB [B,S,C]

  // 1) weights f32 -> f16
  const int wElems = C_ * C_;
  dim3 cvtGrid((wElems + 255) / 256), blk256(256);
  cvt_f32_f16_kernel<<<cvtGrid, blk256, 0, stream>>>(Wq, Wq16, wElems);
  cvt_f32_f16_kernel<<<cvtGrid, blk256, 0, stream>>>(Wk, Wk16, wElems);
  cvt_f32_f16_kernel<<<cvtGrid, blk256, 0, stream>>>(Wv, Wv16, wElems);
  cvt_f32_f16_kernel<<<cvtGrid, blk256, 0, stream>>>(Wo, Wo16, wElems);

  // 2) projections: 64 row-blocks (8 waves each) x 16 col-groups = 1024 blocks
  dim3 gemmGrid((B_ * S_ / (16 * 8)) * (C_ / 64));
  proj_qk_kernel<<<gemmGrid, blk256, 0, stream>>>(query, Wq16, bq, Q16);
  proj_qk_kernel<<<gemmGrid, blk256, 0, stream>>>(key,   Wk16, bk, K16);
  proj_v_kernel<<<gemmGrid, blk256, 0, stream>>>(value, Wv16, bv, V16);

  // 3) attention: 64 (b,h) x 16 q-tile-blocks = 1024 blocks (8 waves each)
  dim3 attnGrid(B_ * H_ * ((S_ / 16) / 8));
  attn_kernel<<<attnGrid, blk256, 0, stream>>>(Q16, K16, V16, O16);

  // 4) output projection (f32 result)
  proj_o_kernel<<<gemmGrid, blk256, 0, stream>>>(O16, Wo16, bo, (float*)d_out);
}
